// RNN_58025008169248
// MI455X (gfx1250) — matrix operand order repaired and checked by reading, not compiled, not run on hardware
//
#include <hip/hip_runtime.h>

typedef __attribute__((ext_vector_type(16))) __bf16 v16bf;
typedef __attribute__((ext_vector_type(8)))  __bf16 v8bf;
typedef __attribute__((ext_vector_type(4)))  __bf16 v4bf;
typedef __attribute__((ext_vector_type(8)))  float  v8f;
typedef __attribute__((ext_vector_type(4)))  float  v4f;

#define RNN_B   256
#define RNN_T   512
#define RNN_DIN 1024
#define RNN_H   1024
#define RNN_K   (RNN_DIN + RNN_H)   // 2048, fused [x_t, h] contraction
#define NBLK    64                   // persistent grid: 16 (N) x 4 (M)
#define LDS_STRIDE 2056              // bf16 elems per W column in LDS (4112 B = 4096 + 16 pad)

static __device__ __forceinline__ v16bf cat8(v8bf a, v8bf b) {
  return __builtin_shufflevector(a, b, 0, 1, 2, 3, 4, 5, 6, 7,
                                       8, 9, 10, 11, 12, 13, 14, 15);
}

// ---------- kernel 0a: W fp32 -> bf16 (4 MB) ----------
__global__ __launch_bounds__(256)
void k_convert_w(const float* __restrict__ W, __bf16* __restrict__ Wbf) {
  size_t i = ((size_t)blockIdx.x * 256 + threadIdx.x) * 4;
  v4f w = *(const v4f*)(W + i);
  v4bf o;
  o.x = (__bf16)w.x; o.y = (__bf16)w.y; o.z = (__bf16)w.z; o.w = (__bf16)w.w;
  *(v4bf*)(Wbf + i) = o;
}

// ---------- kernel 0b: zero h0 + reset grid-barrier counter (every call) ----------
__global__ __launch_bounds__(256)
void k_zero_h(unsigned long long* __restrict__ p, unsigned* __restrict__ sync) {
  p[(size_t)blockIdx.x * 256 + threadIdx.x] = 0ull;
  if (blockIdx.x == 0 && threadIdx.x == 0) *sync = 0u;
}

// ---------- persistent scan kernel ----------
// Wave tile: 16(M) x 32(N). Block: 8 waves = 4M x 2N = 64(M) x 64(N). Grid: 64 blocks.
// Block's 64-column W panel (256 KB bf16) staged in LDS once, reused by all 512 steps.
__global__ __launch_bounds__(256)
void k_rnn_persistent(const float*  __restrict__ xs,    // [B][T][DIN] fp32
                      const __bf16* __restrict__ Wbf,   // [H][K] bf16
                      const float*  __restrict__ bias,  // [H]
                      __bf16*       __restrict__ hA,    // [B][H] ping
                      __bf16*       __restrict__ hB,    // [B][H] pong
                      float*        __restrict__ out,   // [2][B][H] fp32
                      unsigned*     __restrict__ sync)
{
  extern __shared__ char smem_raw[];
  __bf16* smem = (__bf16*)smem_raw;

  const int tid  = threadIdx.x;
  const int lane = tid & 31;
  const int wave = tid >> 5;
  const int half = lane >> 4;     // K-half selector per ISA A/B fragment layouts
  const int l16  = lane & 15;

  const int bx = blockIdx.x & 15;  // N block (64 cols)
  const int by = blockIdx.x >> 4;  // M block (64 rows)
  const int wm = wave >> 1;        // 0..3
  const int wn = wave & 1;         // 0..1
  const int m0 = by * 64 + wm * 16;   // batch rows
  const int n0 = bx * 64 + wn * 32;   // hidden cols

  // ---- one-time: stage this block's W panel [bx*64 .. bx*64+63][0..2047] into LDS ----
  {
    const __bf16* wpan = Wbf + (size_t)(bx * 64) * RNN_K;
    for (int c = tid; c < 64 * 256; c += 256) {   // 16 KB chunks: 64 cols x 256 x 16B
      const int col = c >> 8;
      const int off = (c & 255) * 8;
      v8bf v = *(const v8bf*)(wpan + (size_t)col * RNN_K + off);
      *(v8bf*)(smem + col * LDS_STRIDE + off) = v;
    }
    __syncthreads();
  }

  // A-matrix lane row (16-bit A 16x32: lane = M, VGPR0-3 = K half*8.., VGPR4-7 = K 16+half*8..)
  const int mrow = m0 + l16;
  const float*  xbase = xs + (size_t)mrow * RNN_T * RNN_DIN;
  const __bf16* hArow = hA + (size_t)mrow * RNN_H;
  const __bf16* hBrow = hB + (size_t)mrow * RNN_H;

  // B-matrix from LDS: B[k][n] = W[n][k]; lane l16 = column, K half*8.. and 16+half*8.. per fragment.
  const __bf16* sB0 = smem + (size_t)(wn * 32 + l16) * LDS_STRIDE + (half << 3);
  const __bf16* sB1 = sB0 + 16 * LDS_STRIDE;

  const float b0v = bias[n0 + l16];
  const float b1v = bias[n0 + 16 + l16];
  const int   orow = m0 + (half << 3);

  for (int t = 0; t < RNN_T; ++t) {
    const __bf16* hrow = (t & 1) ? hBrow : hArow;
    __bf16*       hout = (t & 1) ? hA : hB;
    const float*  xrow = xbase + (size_t)t * RNN_DIN;

    // Prefetch next timestep's xs slice (bx==0 blocks, wn==0 waves: covers all 256 rows once).
    if (bx == 0 && wn == 0 && t + 1 < RNN_T) {
      const float* nrow = xrow + RNN_DIN + half * 512;   // each lane-half covers 2 KB
      #pragma unroll
      for (int i = 0; i < 8; ++i)
        __builtin_prefetch(nrow + i * 64, 0, 0);         // global_prefetch_b8, 256B apart
    }

    v8f acc0, acc1;
    #pragma unroll
    for (int r = 0; r < 8; ++r) { acc0[r] = b0v; acc1[r] = b1v; }

    // ---- K 0..1023: x_t contribution (fp32 global loads, on-the-fly bf16 convert) ----
    #pragma unroll 2
    for (int kb = 0; kb < RNN_DIN; kb += 32) {
      const int k0 = kb + half * 8;
      v4f a0 = *(const v4f*)(xrow + k0);
      v4f a1 = *(const v4f*)(xrow + k0 + 4);
      v4f a2 = *(const v4f*)(xrow + k0 + 16);
      v4f a3 = *(const v4f*)(xrow + k0 + 20);
      v16bf af;
      #pragma unroll
      for (int i = 0; i < 4; ++i) {
        af[i]      = (__bf16)a0[i];
        af[4 + i]  = (__bf16)a1[i];
        af[8 + i]  = (__bf16)a2[i];
        af[12 + i] = (__bf16)a3[i];
      }
      v16bf bf0 = cat8(*(const v8bf*)(sB0 + kb), *(const v8bf*)(sB0 + kb + 16));
      v16bf bf1 = cat8(*(const v8bf*)(sB1 + kb), *(const v8bf*)(sB1 + kb + 16));
      acc0 = __builtin_amdgcn_wmma_f32_16x16x32_bf16(false, af, false, bf0,
                                                     (short)0, acc0, false, false);
      acc1 = __builtin_amdgcn_wmma_f32_16x16x32_bf16(false, af, false, bf1,
                                                     (short)0, acc1, false, false);
    }

    // ---- K 1024..2047: recurrent h contribution (bf16 global loads) ----
    #pragma unroll 2
    for (int kb = 0; kb < RNN_H; kb += 32) {
      const int k0 = kb + half * 8;
      v16bf af = cat8(*(const v8bf*)(hrow + k0), *(const v8bf*)(hrow + k0 + 16));
      v16bf bf0 = cat8(*(const v8bf*)(sB0 + RNN_DIN + kb),
                       *(const v8bf*)(sB0 + RNN_DIN + kb + 16));
      v16bf bf1 = cat8(*(const v8bf*)(sB1 + RNN_DIN + kb),
                       *(const v8bf*)(sB1 + RNN_DIN + kb + 16));
      acc0 = __builtin_amdgcn_wmma_f32_16x16x32_bf16(false, af, false, bf0,
                                                     (short)0, acc0, false, false);
      acc1 = __builtin_amdgcn_wmma_f32_16x16x32_bf16(false, af, false, bf1,
                                                     (short)0, acc1, false, false);
    }

    // ---- tanh + store (C/D layout: VGPR r -> M = r + 8*half, N = lane&15) ----
    #pragma unroll
    for (int r = 0; r < 8; ++r) {
      float v0 = tanhf(acc0[r]);
      float v1 = tanhf(acc1[r]);
      size_t o0 = (size_t)(orow + r) * RNN_H + (size_t)(n0 + l16);
      size_t o1 = o0 + 16;
      hout[o0] = (__bf16)v0;
      hout[o1] = (__bf16)v1;
      if (t == RNN_T - 1) {   // final step: fp32 result to both tuple slots
        out[o0] = v0;
        out[o1] = v1;
        out[(size_t)RNN_B * RNN_H + o0] = v0;
        out[(size_t)RNN_B * RNN_H + o1] = v1;
      }
    }

    // ---- device-wide barrier between dependent timesteps (skip after last) ----
    if (t + 1 < RNN_T) {
      __threadfence();          // release this thread's h stores to device scope
      __syncthreads();          // all waves in block arrived
      if (tid == 0) {
        __hip_atomic_fetch_add(sync, 1u, __ATOMIC_ACQ_REL, __HIP_MEMORY_SCOPE_AGENT);
        const unsigned target = (unsigned)(t + 1) * NBLK;
        while (__hip_atomic_load(sync, __ATOMIC_ACQUIRE, __HIP_MEMORY_SCOPE_AGENT) < target)
          __builtin_amdgcn_s_sleep(1);
      }
      __syncthreads();          // fan completion out to all waves
    }
  }
}

extern "C" void kernel_launch(void* const* d_in, const int* in_sizes, int n_in,
                              void* d_out, int out_size, void* d_ws, size_t ws_size,
                              hipStream_t stream) {
  const float* xs   = (const float*)d_in[0];
  const float* W    = (const float*)d_in[1];
  const float* bias = (const float*)d_in[2];
  float* out = (float*)d_out;

  char* ws = (char*)d_ws;
  __bf16* Wbf = (__bf16*)ws;                                            // 4 MB
  __bf16* hA  = (__bf16*)(ws + (size_t)RNN_H * RNN_K * sizeof(__bf16)); // 512 KB
  __bf16* hB  = hA + (size_t)RNN_B * RNN_H;                              // 512 KB
  unsigned* sync = (unsigned*)(hB + (size_t)RNN_B * RNN_H);              // 4 B

  // Per-call setup: convert weights, zero h0, reset barrier counter (deterministic).
  k_convert_w<<<dim3(RNN_H * RNN_K / (256 * 4)), 256, 0, stream>>>(W, Wbf);
  k_zero_h<<<dim3((RNN_B * RNN_H * 2) / (8 * 256)), 256, 0, stream>>>(
      (unsigned long long*)hA, sync);

  // One persistent kernel runs the whole 512-step scan (64 co-resident blocks).
  const size_t lds_bytes = (size_t)64 * LDS_STRIDE * sizeof(__bf16);  // 263,168 B
  k_rnn_persistent<<<dim3(NBLK), 256, lds_bytes, stream>>>(
      xs, Wbf, bias, hA, hB, out, sync);
}
